// SimpleSRNN_6038724018533
// MI455X (gfx1250) — matrix operand order repaired and checked by reading, not compiled, not run on hardware
//
#include <hip/hip_runtime.h>
#include <hip/hip_bf16.h>

// ---------------------------------------------------------------------------
// SRNN (2-layer LIF) for MI455X / gfx1250, compile-only tuning loop.
//   Phase 1: Hx[t,b,:] = x[b,t,:] @ W_in.T + b_in + b_h1   (bulk WMMA GEMM)
//   Phase 2: 512 sequential steps; recurrent GEMMs via WMMA bf16 with
//            TDM (tensor_load_to_lds) double-buffered LDS staging.
// Spikes are exactly representable in bf16; membrane state stays f32.
// This round: hardware bf16 converts (v_cvt_pk_bf16_f32) instead of manual
// RNE bit-twiddle, and hoisted TDM descriptor invariants (less SALU in the
// pipelined chunk loop).
// ---------------------------------------------------------------------------

#define B_   64
#define T_   512
#define DIN_ 512
#define H1_  1024
#define H2_  1024
#define DOUT_ 256
#define BETA_ 0.9f

typedef __attribute__((ext_vector_type(16))) __bf16 v16bf;
typedef __attribute__((ext_vector_type(8)))  float  v8f;
typedef __attribute__((ext_vector_type(4)))  unsigned int u32x4;
typedef __attribute__((ext_vector_type(8)))  int    i32x8;
typedef __attribute__((ext_vector_type(4)))  int    i32x4;

union FragU {
    v16bf  v;
    uint4  q[2];
    __bf16 b[16];
};

__device__ __forceinline__ unsigned short f2bf(float f) {
    __bf16 h = (__bf16)f;                       // hardware cvt (RNE)
    return __builtin_bit_cast(unsigned short, h);
}
__device__ __forceinline__ float bf2f(unsigned short b) {
    unsigned u = ((unsigned)b) << 16;
    return __builtin_bit_cast(float, u);
}

// ---------------------------------------------------------------------------
// TDM 2D tile descriptor (bf16 elements, data_size code 1), global -> LDS.
// D# packing per CDNA5 ISA; groups 2/3 zero => 2D.
// Invariant fields built once; per-issue we patch only lds_addr/global_addr.
// ---------------------------------------------------------------------------
struct Tdm2D {
    u32x4 g0;
    i32x8 g1;

    __device__ __forceinline__ void init(unsigned tensorD0, unsigned tensorD1,
                                         unsigned strideD0,
                                         unsigned tileD0, unsigned tileD1) {
        g0[0] = 1u;                              // count=1 (valid), user mode
        g0[1] = 0u;                              // lds_addr (patched per issue)
        g0[2] = 0u;                              // global_addr lo (patched)
        g0[3] = (2u << 30);                      // global_addr hi | type=2 (patched)
        g1[0] = (int)(1u << 16);                 // wg_mask=0, data_size=1 (2B)
        g1[1] = (int)((tensorD0 & 0xFFFFu) << 16);
        g1[2] = (int)(((tensorD0 >> 16) & 0xFFFFu) | ((tensorD1 & 0xFFFFu) << 16));
        g1[3] = (int)(((tensorD1 >> 16) & 0xFFFFu) | ((tileD0 & 0xFFFFu) << 16));
        g1[4] = (int)(tileD1 & 0xFFFFu);         // tile_dim1, tile_dim2=0
        g1[5] = (int)strideD0;                   // tensor_dim0_stride lo
        g1[6] = 0;
        g1[7] = 0;
    }

    __device__ __forceinline__ void issue(unsigned ldsOff, const void* gptr) {
        unsigned long long ga = (unsigned long long)gptr;
        g0[1] = ldsOff;
        g0[2] = (unsigned)(ga & 0xFFFFFFFFu);
        g0[3] = (unsigned)((ga >> 32) & 0x01FFFFFFu) | (2u << 30);
        i32x4 z4 = {0, 0, 0, 0};
#if defined(__clang_major__) && (__clang_major__ >= 23)
        i32x8 z8 = {0, 0, 0, 0, 0, 0, 0, 0};
        __builtin_amdgcn_tensor_load_to_lds(g0, g1, z4, z4, z8, 0);
#else
        __builtin_amdgcn_tensor_load_to_lds(g0, g1, z4, z4, 0);
#endif
    }
};

// ---------------------------------------------------------------------------
// Recurrent-phase GEMM building block:
//   acc(16x16 f32 tile) += A[64xK] @ W[N x K]^T restricted to 16 output cols
// Block = 128 threads = 4 waves; wave w owns batch rows [16w,16w+16).
// K-chunked (128) TDM double-buffered LDS staging; wave0 drives the DMA.
// ---------------------------------------------------------------------------
__device__ __forceinline__ void mm_pair(const unsigned short* __restrict__ Wrow0, // W[n0][0], row stride K
                                        const unsigned short* __restrict__ A,     // [64][K]
                                        int K, v8f& acc, int lane, int wave)
{
    constexpr int CH = 128;
    __shared__ unsigned short sW[2][16 * CH];
    __shared__ unsigned short sA[2][64 * CH];
    const int nch = K / CH;
    const int h  = lane >> 4;
    const int rA = (wave << 4) + (lane & 15);   // batch row for A fragment
    const int rB = lane & 15;                   // strip row for B fragment

    if (wave == 0) {
        Tdm2D dW, dA;
        dW.init((unsigned)K, 16u, (unsigned)K, CH, 16u);
        dA.init((unsigned)K, 64u, (unsigned)K, CH, 64u);
        dW.issue((unsigned)(unsigned long long)&sW[0][0], Wrow0);
        dA.issue((unsigned)(unsigned long long)&sA[0][0], A);
        for (int c = 0; c < nch; ++c) {
            const int cur = c & 1;
            if (c + 1 < nch) {
                // prefetch next chunk into the other buffer, then wait for
                // chunk c (TDM is in-order per wave: cnt<=2 => first pair done)
                dW.issue((unsigned)(unsigned long long)&sW[cur ^ 1][0],
                         Wrow0 + (size_t)(c + 1) * CH);
                dA.issue((unsigned)(unsigned long long)&sA[cur ^ 1][0],
                         A + (size_t)(c + 1) * CH);
                __builtin_amdgcn_s_wait_tensorcnt(2);
            } else {
                __builtin_amdgcn_s_wait_tensorcnt(0);
            }
            __syncthreads();
#pragma unroll
            for (int ks = 0; ks < CH / 32; ++ks) {
                FragU a, b;
                const unsigned short* pa = &sA[cur][rA * CH + ks * 32 + h * 8];
                a.q[0] = *(const uint4*)(pa);
                a.q[1] = *(const uint4*)(pa + 16);
                const unsigned short* pb = &sW[cur][rB * CH + ks * 32 + h * 8];
                b.q[0] = *(const uint4*)(pb);
                b.q[1] = *(const uint4*)(pb + 16);
                acc = __builtin_amdgcn_wmma_f32_16x16x32_bf16(
                    false, a.v, false, b.v, (short)0, acc, false, false);
            }
            __syncthreads();
        }
    } else {
        for (int c = 0; c < nch; ++c) {
            const int cur = c & 1;
            __syncthreads();
#pragma unroll
            for (int ks = 0; ks < CH / 32; ++ks) {
                FragU a, b;
                const unsigned short* pa = &sA[cur][rA * CH + ks * 32 + h * 8];
                a.q[0] = *(const uint4*)(pa);
                a.q[1] = *(const uint4*)(pa + 16);
                const unsigned short* pb = &sW[cur][rB * CH + ks * 32 + h * 8];
                b.q[0] = *(const uint4*)(pb);
                b.q[1] = *(const uint4*)(pb + 16);
                acc = __builtin_amdgcn_wmma_f32_16x16x32_bf16(
                    false, a.v, false, b.v, (short)0, acc, false, false);
            }
            __syncthreads();
        }
    }
}

// ---------------------------------------------------------------------------
// Phase-1 GEMM: Hx[t][b][n] = bf16( x[b][t][:] @ W_in[n][:] + b_in[n] + b_h1[n] )
// One wave per 16x16 tile, K=512 streamed from global with f32->bf16 convert.
// ---------------------------------------------------------------------------
__global__ __launch_bounds__(256) void gemm_in_kernel(
    const float* __restrict__ x,      // [B][T][DIN]
    const float* __restrict__ Win,    // [H1][DIN]
    const float* __restrict__ b_in,
    const float* __restrict__ b_h1,
    unsigned short* __restrict__ Hx)  // [T][B][H1] bf16
{
    const int lane = threadIdx.x & 31, wave = threadIdx.x >> 5;
    const int tile = blockIdx.x * 8 + wave;       // (B*T/16) * (H1/16) tiles
    const int nt = tile & 63, mt = tile >> 6;
    const int n0 = nt << 4, r0 = mt << 4;
    const int h = lane >> 4;
    const int r = r0 + (lane & 15);               // row in [B*T): r = b*T + t
    const int n = n0 + (lane & 15);

    v8f acc = {0.f, 0.f, 0.f, 0.f, 0.f, 0.f, 0.f, 0.f};
    for (int kc = 0; kc < DIN_; kc += 32) {
        FragU a, b;
        const float* px = &x[(size_t)r * DIN_ + kc + h * 8];
        if (kc + 32 < DIN_) __builtin_prefetch(px + 32, 0, 0);   // global_prefetch_b8
        float4 xa0 = *(const float4*)(px);
        float4 xa1 = *(const float4*)(px + 4);
        float4 xa2 = *(const float4*)(px + 16);
        float4 xa3 = *(const float4*)(px + 20);
        a.b[0]  = (__bf16)xa0.x; a.b[1]  = (__bf16)xa0.y;
        a.b[2]  = (__bf16)xa0.z; a.b[3]  = (__bf16)xa0.w;
        a.b[4]  = (__bf16)xa1.x; a.b[5]  = (__bf16)xa1.y;
        a.b[6]  = (__bf16)xa1.z; a.b[7]  = (__bf16)xa1.w;
        a.b[8]  = (__bf16)xa2.x; a.b[9]  = (__bf16)xa2.y;
        a.b[10] = (__bf16)xa2.z; a.b[11] = (__bf16)xa2.w;
        a.b[12] = (__bf16)xa3.x; a.b[13] = (__bf16)xa3.y;
        a.b[14] = (__bf16)xa3.z; a.b[15] = (__bf16)xa3.w;
        const float* pw = &Win[(size_t)n * DIN_ + kc + h * 8];
#pragma unroll
        for (int j = 0; j < 4; ++j) {
            float4 w4 = *(const float4*)(pw + (j & 1) * 4 + (j >> 1) * 16);
            b.b[4 * j + 0] = (__bf16)w4.x;
            b.b[4 * j + 1] = (__bf16)w4.y;
            b.b[4 * j + 2] = (__bf16)w4.z;
            b.b[4 * j + 3] = (__bf16)w4.w;
        }
        acc = __builtin_amdgcn_wmma_f32_16x16x32_bf16(
            false, a.v, false, b.v, (short)0, acc, false, false);
    }
    const float bias = b_in[n] + b_h1[n];
#pragma unroll
    for (int i = 0; i < 8; ++i) {
        const int rr = r0 + (h << 3) + i;
        const int t  = rr & (T_ - 1);
        const int bb = rr >> 9;                   // rr / T_
        Hx[((size_t)t * B_ + bb) * H1_ + n] = f2bf(acc[i] + bias);
    }
}

// ---------------------------------------------------------------------------
// Per-step kernels (grid = N/16 blocks, 128 threads = 4 M-tiles)
// ---------------------------------------------------------------------------
__global__ __launch_bounds__(128) void layer1_kernel(
    const unsigned short* __restrict__ Hx_t,  // [B][H1] bf16 (offset by t)
    const unsigned short* __restrict__ W1b,   // [H1][H1] bf16
    const unsigned short* __restrict__ s1r,   // [B][H1] bf16 (read)
    unsigned short* __restrict__ s1w,         // [B][H1] bf16 (write)
    float* __restrict__ v1)                   // [B][H1] f32 (in place)
{
    const int lane = threadIdx.x & 31, wave = threadIdx.x >> 5;
    const int n0 = blockIdx.x << 4;
    v8f acc = {0.f, 0.f, 0.f, 0.f, 0.f, 0.f, 0.f, 0.f};
    mm_pair(W1b + (size_t)n0 * H1_, s1r, H1_, acc, lane, wave);

    const int h = lane >> 4;
    const int n = n0 + (lane & 15);
#pragma unroll
    for (int i = 0; i < 8; ++i) {
        const int bidx = (wave << 4) + (h << 3) + i;
        const size_t idx = (size_t)bidx * H1_ + n;
        const float hin = bf2f(Hx_t[idx]) + acc[i];
        const float v = BETA_ * v1[idx] + hin;
        const bool fire = (v >= 1.0f);
        v1[idx] = fire ? (v - 1.0f) : v;
        s1w[idx] = fire ? (unsigned short)0x3F80 : (unsigned short)0;  // bf16 1.0/0.0
    }
}

__global__ __launch_bounds__(128) void layer2_kernel(
    const float* __restrict__ bias2,          // [H2] = b_12 + b_h2
    const unsigned short* __restrict__ W12b,  // [H2][H1] bf16
    const unsigned short* __restrict__ s1n,   // new s1 (this step)
    const unsigned short* __restrict__ Wh2b,  // [H2][H2] bf16
    const unsigned short* __restrict__ s2r,
    unsigned short* __restrict__ s2w,
    float* __restrict__ v2)
{
    const int lane = threadIdx.x & 31, wave = threadIdx.x >> 5;
    const int n0 = blockIdx.x << 4;
    v8f acc = {0.f, 0.f, 0.f, 0.f, 0.f, 0.f, 0.f, 0.f};
    mm_pair(W12b + (size_t)n0 * H1_, s1n, H1_, acc, lane, wave);
    mm_pair(Wh2b + (size_t)n0 * H2_, s2r, H2_, acc, lane, wave);

    const int h = lane >> 4;
    const int n = n0 + (lane & 15);
    const float bias = bias2[n];
#pragma unroll
    for (int i = 0; i < 8; ++i) {
        const int bidx = (wave << 4) + (h << 3) + i;
        const size_t idx = (size_t)bidx * H2_ + n;
        const float v = BETA_ * v2[idx] + acc[i] + bias;
        const bool fire = (v >= 1.0f);
        v2[idx] = fire ? (v - 1.0f) : v;
        s2w[idx] = fire ? (unsigned short)0x3F80 : (unsigned short)0;
    }
}

__global__ __launch_bounds__(128) void readout_kernel(
    const unsigned short* __restrict__ Woutb, // [DOUT][H2] bf16
    const unsigned short* __restrict__ s2n,
    float* __restrict__ out)                  // [B][DOUT], accumulated
{
    const int lane = threadIdx.x & 31, wave = threadIdx.x >> 5;
    const int n0 = blockIdx.x << 4;
    v8f acc = {0.f, 0.f, 0.f, 0.f, 0.f, 0.f, 0.f, 0.f};
    mm_pair(Woutb + (size_t)n0 * H2_, s2n, H2_, acc, lane, wave);

    const int h = lane >> 4;
    const int n = n0 + (lane & 15);
#pragma unroll
    for (int i = 0; i < 8; ++i) {
        const int bidx = (wave << 4) + (h << 3) + i;
        out[(size_t)bidx * DOUT_ + n] += acc[i];   // single writer per element
    }
}

// ---------------------------------------------------------------------------
// Prep kernels
// ---------------------------------------------------------------------------
__global__ void cvt_bf16_kernel(const float* __restrict__ src,
                                unsigned short* __restrict__ dst, int n2) {
    // two elements per thread so the compiler can use packed bf16 converts
    int i = (blockIdx.x * blockDim.x + threadIdx.x) * 2;
    if (i < n2) {
        dst[i]     = f2bf(src[i]);
        dst[i + 1] = f2bf(src[i + 1]);
    }
}

__global__ void bias2_kernel(const float* __restrict__ b12,
                             const float* __restrict__ bh2,
                             float* __restrict__ bias2) {
    int i = blockIdx.x * blockDim.x + threadIdx.x;
    if (i < H2_) bias2[i] = b12[i] + bh2[i];
}

__global__ void seed_out_kernel(const float* __restrict__ b_out,
                                float* __restrict__ out) {
    int i = blockIdx.x * blockDim.x + threadIdx.x;
    if (i < B_ * DOUT_) out[i] = (float)T_ * b_out[i & (DOUT_ - 1)];
}

// ---------------------------------------------------------------------------
extern "C" void kernel_launch(void* const* d_in, const int* in_sizes, int n_in,
                              void* d_out, int out_size, void* d_ws, size_t ws_size,
                              hipStream_t stream) {
    (void)in_sizes; (void)n_in; (void)out_size; (void)ws_size;
    const float* x     = (const float*)d_in[0];
    const float* W_in  = (const float*)d_in[1];
    const float* b_in  = (const float*)d_in[2];
    const float* W_h1  = (const float*)d_in[3];
    const float* b_h1  = (const float*)d_in[4];
    const float* W_12  = (const float*)d_in[5];
    const float* b_12  = (const float*)d_in[6];
    const float* W_h2  = (const float*)d_in[7];
    const float* b_h2  = (const float*)d_in[8];
    const float* W_out = (const float*)d_in[9];
    const float* b_out = (const float*)d_in[10];
    float* out = (float*)d_out;

    char* ws = (char*)d_ws;
    size_t off = 0;
    auto take = [&](size_t bytes) -> char* {
        char* p = ws + off;
        off = (off + bytes + 255) & ~(size_t)255;
        return p;
    };
    unsigned short* Hx    = (unsigned short*)take((size_t)T_ * B_ * H1_ * 2);  // 64 MB
    unsigned short* W1b   = (unsigned short*)take((size_t)H1_ * H1_ * 2);
    unsigned short* W12b  = (unsigned short*)take((size_t)H2_ * H1_ * 2);
    unsigned short* Wh2b  = (unsigned short*)take((size_t)H2_ * H2_ * 2);
    unsigned short* Woutb = (unsigned short*)take((size_t)DOUT_ * H2_ * 2);
    unsigned short* s1a   = (unsigned short*)take((size_t)B_ * H1_ * 2);
    unsigned short* s1b   = (unsigned short*)take((size_t)B_ * H1_ * 2);
    unsigned short* s2a   = (unsigned short*)take((size_t)B_ * H2_ * 2);
    unsigned short* s2b   = (unsigned short*)take((size_t)B_ * H2_ * 2);
    float* v1    = (float*)take((size_t)B_ * H1_ * 4);
    float* v2    = (float*)take((size_t)B_ * H2_ * 4);
    float* bias2 = (float*)take((size_t)H2_ * 4);

    // init state (graph-capture-safe)
    hipMemsetAsync(v1, 0, (size_t)B_ * H1_ * 4, stream);
    hipMemsetAsync(v2, 0, (size_t)B_ * H2_ * 4, stream);
    hipMemsetAsync(s1a, 0, (size_t)B_ * H1_ * 2, stream);
    hipMemsetAsync(s2a, 0, (size_t)B_ * H2_ * 2, stream);

    // weight conversion to bf16 (one-time; L2-resident thereafter)
    cvt_bf16_kernel<<<(H1_ * H1_ / 2 + 255) / 256, 256, 0, stream>>>(W_h1, W1b, H1_ * H1_);
    cvt_bf16_kernel<<<(H2_ * H1_ / 2 + 255) / 256, 256, 0, stream>>>(W_12, W12b, H2_ * H1_);
    cvt_bf16_kernel<<<(H2_ * H2_ / 2 + 255) / 256, 256, 0, stream>>>(W_h2, Wh2b, H2_ * H2_);
    cvt_bf16_kernel<<<(DOUT_ * H2_ / 2 + 255) / 256, 256, 0, stream>>>(W_out, Woutb, DOUT_ * H2_);
    bias2_kernel<<<(H2_ + 255) / 256, 256, 0, stream>>>(b_12, b_h2, bias2);
    seed_out_kernel<<<(B_ * DOUT_ + 255) / 256, 256, 0, stream>>>(b_out, out);

    // Phase 1: bulk input GEMM, hoisted out of the scan.
    // (B*T/16)=2048 M-tiles x (H1/16)=64 N-tiles, 8 wave-tiles per block.
    gemm_in_kernel<<<(2048 * 64) / 8, 256, 0, stream>>>(x, W_in, b_in, b_h1, Hx);

    // Phase 2: sequential scan; double-buffered spike arrays by parity.
    for (int t = 0; t < T_; ++t) {
        const unsigned short* s1r = (t & 1) ? s1b : s1a;
        unsigned short*       s1w = (t & 1) ? s1a : s1b;
        const unsigned short* s2r = (t & 1) ? s2b : s2a;
        unsigned short*       s2w = (t & 1) ? s2a : s2b;
        const unsigned short* Hx_t = Hx + (size_t)t * B_ * H1_;

        layer1_kernel<<<H1_ / 16, 128, 0, stream>>>(Hx_t, W1b, s1r, s1w, v1);
        layer2_kernel<<<H2_ / 16, 128, 0, stream>>>(bias2, W12b, s1w, Wh2b, s2r, s2w, v2);
        readout_kernel<<<DOUT_ / 16, 128, 0, stream>>>(Woutb, s2w, out);
    }
}
